// PaaSModel_19670950216273
// MI455X (gfx1250) — matrix operand order repaired and checked by reading, not compiled, not run on hardware
//
#include <hip/hip_runtime.h>
#include <hip/hip_bf16.h>
#include <cstdint>

// ---------------- constants (reference shapes) ----------------
#define HASHSZ   21000
#define NUM_LT   6
#define NUM_GT   5
#define NUM_EMB  12
#define NUM_LABEL 6
#define EDIM     128
#define BATCH    8192
#define BAGL     20
#define H1DIM    1024
#define H2DIM    512
#define D0       1614            // 78 + 12*128
#define D0P      1632            // padded to multiple of 32 for WMMA K
#define NPAIR    78

typedef __attribute__((ext_vector_type(16))) __bf16 v16bf;
typedef __attribute__((ext_vector_type(8)))  float  v8f;

#if __has_builtin(__builtin_amdgcn_global_load_async_to_lds_b128)
#define USE_ASYNC_LDS 1
#else
#define USE_ASYNC_LDS 0
#endif

__device__ __forceinline__ uint16_t f2bf(float f) {
  uint32_t u = __builtin_bit_cast(uint32_t, f);
  u += 0x7FFFu + ((u >> 16) & 1u);           // round-to-nearest-even
  return (uint16_t)(u >> 16);
}

#if USE_ASYNC_LDS
typedef int v4i __attribute__((vector_size(16)));
typedef __attribute__((address_space(1))) v4i* as1_v4i;
typedef __attribute__((address_space(3))) v4i* as3_v4i;
__device__ __forceinline__ void async_b128(const void* g, void* l) {
  __builtin_amdgcn_global_load_async_to_lds_b128((as1_v4i)g, (as3_v4i)l, 0, 0);
}
__device__ __forceinline__ void wait_async_cur() {   // current tile done, next in flight
#if __has_builtin(__builtin_amdgcn_s_wait_asynccnt)
  __builtin_amdgcn_s_wait_asynccnt(4);
#else
  asm volatile("s_wait_asynccnt 0x4" ::: "memory");
#endif
}
__device__ __forceinline__ void wait_async_all() {
#if __has_builtin(__builtin_amdgcn_s_wait_asynccnt)
  __builtin_amdgcn_s_wait_asynccnt(0);
#else
  asm volatile("s_wait_asynccnt 0x0" ::: "memory");
#endif
}
#endif

// ---------------- weight convert + K-pad to bf16 ----------------
__global__ void k_cvt_w(const float* __restrict__ W, uint16_t* __restrict__ Wb,
                        int N, int Kin, int Kout) {
  int idx = blockIdx.x * 256 + threadIdx.x;
  if (idx >= N * Kout) return;
  int n = idx / Kout, k = idx - n * Kout;
  float v = (k < Kin) ? W[(size_t)n * Kin + k] : 0.0f;
  Wb[idx] = f2bf(v);
}

// ---------------- embedding-bag(max) + interaction + assemble 'over' row ----
__global__ __launch_bounds__(128)
void k_embed(const int* __restrict__ lt_idx,  // [6,B,20]
             const int* __restrict__ gt_idx,  // [5,B,20]
             const int* __restrict__ show_ids,
             const float* __restrict__ lt_tab,   // [6,HASH,128]
             const float* __restrict__ gt_tab,   // [5,HASH,128]
             const float* __restrict__ show_tab, // [HASH,128]
             uint16_t* __restrict__ over)        // [B, D0P] bf16
{
  const int b = blockIdx.x;
  const int e = threadIdx.x;                 // 0..127
  __shared__ float feats[NUM_EMB][EDIM];

  for (int t = 0; t < NUM_LT; ++t) {
    const int*   ip  = lt_idx + ((size_t)t * BATCH + b) * BAGL;
    const float* tab = lt_tab + (size_t)t * HASHSZ * EDIM;
    float m = -INFINITY;
    #pragma unroll 4
    for (int l = 0; l < BAGL; ++l)
      m = fmaxf(m, tab[(size_t)ip[l] * EDIM + e]);
    feats[t][e] = m;
  }
  for (int t = 0; t < NUM_GT; ++t) {
    const int*   ip  = gt_idx + ((size_t)t * BATCH + b) * BAGL;
    const float* tab = gt_tab + (size_t)t * HASHSZ * EDIM;
    float m = -INFINITY;
    #pragma unroll 4
    for (int l = 0; l < BAGL; ++l)
      m = fmaxf(m, tab[(size_t)ip[l] * EDIM + e]);
    feats[NUM_LT + t][e] = m;
  }
  feats[NUM_EMB - 1][e] = show_tab[(size_t)show_ids[b] * EDIM + e];
  __syncthreads();

  uint16_t* row = over + (size_t)b * D0P;
  // cats: cols 78 .. 78+12*128-1
  for (int t = 0; t < NUM_EMB; ++t)
    row[NPAIR + t * EDIM + e] = f2bf(feats[t][e]);
  // zero K-pad: cols 1614..1631
  if (e < D0P - D0) row[D0 + e] = 0;

  // dots: 78 upper-tri pairs, 4 waves x 32 lanes
  const int wid = e >> 5, lane = e & 31;
  for (int p = wid; p < NPAIR; p += 4) {
    int i = 0, rem = p;
    while (rem >= NUM_EMB - i) { rem -= NUM_EMB - i; ++i; }
    const int j = i + rem;
    float s = 0.0f;
    #pragma unroll
    for (int c = lane; c < EDIM; c += 32) s += feats[i][c] * feats[j][c];
    #pragma unroll
    for (int off = 16; off; off >>= 1) s += __shfl_xor(s, off, 32);
    if (lane == 0) row[p] = f2bf(s * (1.0f / (float)EDIM));
  }
}

// ---------------- bf16 WMMA GEMM:  Y[M,N] = X[M,K] * W[N,K]^T (f32 accum) ---
#define BM 128
#define BN 128
#define BK 32
#define LDSW 40   // padded LDS row stride (bf16 elems) to spread banks

// one K-step of WMMA work from a staged LDS tile pair
__device__ __forceinline__ void tile_compute(const uint16_t* __restrict__ Asb,
                                             const uint16_t* __restrict__ Bsb,
                                             int wm, int wn, int lane,
                                             v8f (&acc)[2][4]) {
  const int l15 = lane & 15;
  // A fragment: lane = M%16; K-octet base 0 (lanes<16) / 8 (lanes>=16), +16
  const int abase = (lane < 16) ? 0 : 8;
  v16bf afrag[2];
  #pragma unroll
  for (int i = 0; i < 2; ++i) {
    const uint16_t* ap = &Asb[(wm + i * 16 + l15) * LDSW + abase];
    union { v16bf v; uint4 q[2]; } u;
    u.q[0] = *(const uint4*)ap;
    u.q[1] = *(const uint4*)(ap + 16);
    afrag[i] = u.v;
  }
  // B fragment: lane%16 = N col; K-octet base 0 (lanes<16) / 8 (lanes>=16), +16
  const int bbase = (lane < 16) ? 0 : 8;
  v16bf bfrag[4];
  #pragma unroll
  for (int j = 0; j < 4; ++j) {
    const uint16_t* bp = &Bsb[(wn + j * 16 + l15) * LDSW + bbase];
    union { v16bf v; uint4 q[2]; } u;
    u.q[0] = *(const uint4*)bp;
    u.q[1] = *(const uint4*)(bp + 16);
    bfrag[j] = u.v;
  }
  #pragma unroll
  for (int i = 0; i < 2; ++i)
    #pragma unroll
    for (int j = 0; j < 4; ++j)
      acc[i][j] = __builtin_amdgcn_wmma_f32_16x16x32_bf16(
          false, afrag[i], false, bfrag[j], (short)0, acc[i][j], false, false);
}

__global__ __launch_bounds__(256)
void k_gemm_bf16(const uint16_t* __restrict__ X, const uint16_t* __restrict__ W,
                 float* __restrict__ Y, int M, int N, int K) {
  const int tid  = threadIdx.x;
  const int bm   = blockIdx.x * BM;
  const int bn   = blockIdx.y * BN;
  const int wid  = tid >> 5;
  const int lane = tid & 31;
  const int wm   = (wid & 3) * 32;   // wave's M offset inside tile (2 x 16)
  const int wn   = (wid >> 2) * 64;  // wave's N offset inside tile (4 x 16)
  const int l15  = lane & 15;

  v8f acc[2][4] = {};

  // cooperative tile loaders: 256 thr, each one row-half (16 bf16 = 2 x b128)
  const int lr = tid >> 1;
  const int lc = (tid & 1) * 16;
  const uint16_t* xg = X + (size_t)(bm + lr) * K + lc;
  const uint16_t* wg = W + (size_t)(bn + lr) * K + lc;

#if USE_ASYNC_LDS
  // double-buffered tiles, filled by async DMA (no VGPR staging)
  __shared__ __align__(16) uint16_t As[2][BM * LDSW];
  __shared__ __align__(16) uint16_t Bs[2][BN * LDSW];
  const int ntiles = K / BK;
  const int ao = lr * LDSW + lc;

  // prologue: tile 0 -> buffer 0   (4 async b128 per wave)
  async_b128(xg,     &As[0][ao]);
  async_b128(xg + 8, &As[0][ao + 8]);
  async_b128(wg,     &Bs[0][ao]);
  async_b128(wg + 8, &Bs[0][ao + 8]);

  for (int kt = 0; kt < ntiles; ++kt) {
    const int buf = kt & 1;
    if (kt + 1 < ntiles) {
      const uint16_t* xn = xg + (size_t)(kt + 1) * BK;
      const uint16_t* wn_ = wg + (size_t)(kt + 1) * BK;
      async_b128(xn,      &As[buf ^ 1][ao]);
      async_b128(xn + 8,  &As[buf ^ 1][ao + 8]);
      async_b128(wn_,     &Bs[buf ^ 1][ao]);
      async_b128(wn_ + 8, &Bs[buf ^ 1][ao + 8]);
      wait_async_cur();            // this wave's tile-kt DMAs retired (in order)
    } else {
      wait_async_all();
    }
    __syncthreads();               // all waves' DMAs for tile kt visible
    tile_compute(As[buf], Bs[buf], wm, wn, lane, acc);
    __syncthreads();               // tile kt fully consumed before refill
  }
#else
  // fallback: synchronous staging through VGPRs, single buffer
  __shared__ __align__(16) uint16_t As[BM * LDSW];
  __shared__ __align__(16) uint16_t Bs[BN * LDSW];
  for (int k0 = 0; k0 < K; k0 += BK) {
    uint4 xa = *(const uint4*)(xg + k0);
    uint4 xb = *(const uint4*)(xg + k0 + 8);
    uint4 wa = *(const uint4*)(wg + k0);
    uint4 wb = *(const uint4*)(wg + k0 + 8);
    __syncthreads();
    *(uint4*)&As[lr * LDSW + lc]     = xa;
    *(uint4*)&As[lr * LDSW + lc + 8] = xb;
    *(uint4*)&Bs[lr * LDSW + lc]     = wa;
    *(uint4*)&Bs[lr * LDSW + lc + 8] = wb;
    __syncthreads();
    tile_compute(As, Bs, wm, wn, lane, acc);
  }
  __syncthreads();
#endif

  // store: C/D layout -> m = i*16 + r + 8*(lane>=16), n = lane%16
  const int mhi = 8 * (lane >> 4);
  #pragma unroll
  for (int i = 0; i < 2; ++i) {
    #pragma unroll
    for (int j = 0; j < 4; ++j) {
      const int ncol = bn + wn + j * 16 + l15;
      #pragma unroll
      for (int r = 0; r < 8; ++r) {
        const int mrow = bm + wm + i * 16 + mhi + r;
        Y[(size_t)mrow * N + ncol] = acc[i][j][r];
      }
    }
  }
}

// ---------------- batch stats -> fused scale/shift (bias-free: b cancels) ---
__global__ __launch_bounds__(256)
void k_bnstats(const float* __restrict__ H, const float* __restrict__ g,
               const float* __restrict__ be, float* __restrict__ scale,
               float* __restrict__ shift, int Bn, int N) {
  const int cx = threadIdx.x & 63;
  const int ry = threadIdx.x >> 6;          // 0..3
  const int n  = blockIdx.x * 64 + cx;
  float s = 0.0f, s2 = 0.0f;
  for (int r = ry; r < Bn; r += 4) {
    float v = H[(size_t)r * N + n];
    s += v; s2 += v * v;
  }
  __shared__ float sh[2][4][64];
  sh[0][ry][cx] = s; sh[1][ry][cx] = s2;
  __syncthreads();
  if (ry == 0) {
    s  = sh[0][0][cx] + sh[0][1][cx] + sh[0][2][cx] + sh[0][3][cx];
    s2 = sh[1][0][cx] + sh[1][1][cx] + sh[1][2][cx] + sh[1][3][cx];
    const float inv = 1.0f / (float)Bn;
    const float mu  = s * inv;
    const float var = s2 * inv - mu * mu;    // biased var, matches reference
    const float sc  = g[n] * rsqrtf(var + 1e-5f);
    scale[n] = sc;
    shift[n] = be[n] - mu * sc;
  }
}

// ---------------- apply BN + SiLU, emit bf16 (next GEMM) or f32 (head) ------
template <bool BF16OUT>
__global__ void k_bnact(const float* __restrict__ H, const float* __restrict__ scale,
                        const float* __restrict__ shift, void* __restrict__ out,
                        int total, int N) {
  int idx = blockIdx.x * 256 + threadIdx.x;
  if (idx >= total) return;
  int n = idx % N;
  float h = H[idx] * scale[n] + shift[n];
  float y = h / (1.0f + __expf(-h));         // SiLU
  if (BF16OUT) ((uint16_t*)out)[idx] = f2bf(y);
  else         ((float*)out)[idx]    = y;
}

// ---------------- head: out[l][b] = X3[b,:].head_W[l,:] + head_b[l] ---------
__global__ __launch_bounds__(256)
void k_head(const float* __restrict__ X, const float* __restrict__ hw,
            const float* __restrict__ hb, float* __restrict__ out, int Bn) {
  __shared__ float w[NUM_LABEL * H2DIM];
  for (int i = threadIdx.x; i < NUM_LABEL * H2DIM; i += 256) w[i] = hw[i];
  __syncthreads();
  const int b = blockIdx.x * 256 + threadIdx.x;
  if (b >= Bn) return;
  const float* x = X + (size_t)b * H2DIM;
  float acc[NUM_LABEL];
  #pragma unroll
  for (int l = 0; l < NUM_LABEL; ++l) acc[l] = hb[l];
  for (int k = 0; k < H2DIM; ++k) {
    const float xv = x[k];
    #pragma unroll
    for (int l = 0; l < NUM_LABEL; ++l) acc[l] += xv * w[l * H2DIM + k];
  }
  #pragma unroll
  for (int l = 0; l < NUM_LABEL; ++l) out[(size_t)l * Bn + b] = acc[l];
}

// ---------------- host orchestration ----------------------------------------
extern "C" void kernel_launch(void* const* d_in, const int* in_sizes, int n_in,
                              void* d_out, int out_size, void* d_ws, size_t ws_size,
                              hipStream_t stream) {
  const int*   lt_idx   = (const int*)  d_in[0];
  const int*   gt_idx   = (const int*)  d_in[1];
  const int*   show_ids = (const int*)  d_in[2];
  const float* lt_tab   = (const float*)d_in[3];
  const float* gt_tab   = (const float*)d_in[4];
  const float* show_tab = (const float*)d_in[5];
  const float* W1  = (const float*)d_in[6];
  // b1 = d_in[7]  (provably cancels in BatchNorm)
  const float* g1  = (const float*)d_in[8];
  const float* be1 = (const float*)d_in[9];
  const float* W2  = (const float*)d_in[10];
  // b2 = d_in[11] (cancels)
  const float* g2  = (const float*)d_in[12];
  const float* be2 = (const float*)d_in[13];
  const float* hw  = (const float*)d_in[14];
  const float* hb  = (const float*)d_in[15];
  float* out = (float*)d_out;

  // workspace layout (256B aligned slices)
  char* ws = (char*)d_ws;
  size_t off = 0;
  auto take = [&](size_t bytes) { char* p = ws + off; off = (off + bytes + 255) & ~(size_t)255; return p; };
  uint16_t* overB = (uint16_t*)take((size_t)BATCH * D0P * 2);
  uint16_t* W1B   = (uint16_t*)take((size_t)H1DIM * D0P * 2);
  uint16_t* W2B   = (uint16_t*)take((size_t)H2DIM * H1DIM * 2);
  float*    Hh1   = (float*)   take((size_t)BATCH * H1DIM * 4);
  uint16_t* X2B   = (uint16_t*)take((size_t)BATCH * H1DIM * 2);
  float*    Hh2   = (float*)   take((size_t)BATCH * H2DIM * 4);
  float*    X3F   = (float*)   take((size_t)BATCH * H2DIM * 4);
  float*    sc1   = (float*)   take(H1DIM * 4);
  float*    sh1   = (float*)   take(H1DIM * 4);
  float*    sc2   = (float*)   take(H2DIM * 4);
  float*    sh2   = (float*)   take(H2DIM * 4);
  (void)ws_size; (void)in_sizes; (void)n_in; (void)out_size;

  // 1) weights -> bf16 (K-padded)
  k_cvt_w<<<dim3((H1DIM * D0P + 255) / 256), dim3(256), 0, stream>>>(W1, W1B, H1DIM, D0, D0P);
  k_cvt_w<<<dim3((H2DIM * H1DIM + 255) / 256), dim3(256), 0, stream>>>(W2, W2B, H2DIM, H1DIM, H1DIM);

  // 2) embeddings + interaction -> over (bf16)
  k_embed<<<dim3(BATCH), dim3(128), 0, stream>>>(lt_idx, gt_idx, show_ids,
                                                 lt_tab, gt_tab, show_tab, overB);

  // 3) GEMM1: [8192,1632] x [1024,1632]^T -> H1
  k_gemm_bf16<<<dim3(BATCH / BM, H1DIM / BN), dim3(256), 0, stream>>>(
      overB, W1B, Hh1, BATCH, H1DIM, D0P);

  // 4) BN1 stats + apply + SiLU -> bf16 X2
  k_bnstats<<<dim3(H1DIM / 64), dim3(256), 0, stream>>>(Hh1, g1, be1, sc1, sh1, BATCH, H1DIM);
  k_bnact<true><<<dim3((BATCH * H1DIM + 255) / 256), dim3(256), 0, stream>>>(
      Hh1, sc1, sh1, (void*)X2B, BATCH * H1DIM, H1DIM);

  // 5) GEMM2: [8192,1024] x [512,1024]^T -> H2
  k_gemm_bf16<<<dim3(BATCH / BM, H2DIM / BN), dim3(256), 0, stream>>>(
      X2B, W2B, Hh2, BATCH, H2DIM, H1DIM);

  // 6) BN2 stats + apply + SiLU -> f32 X3
  k_bnstats<<<dim3(H2DIM / 64), dim3(256), 0, stream>>>(Hh2, g2, be2, sc2, sh2, BATCH, H2DIM);
  k_bnact<false><<<dim3((BATCH * H2DIM + 255) / 256), dim3(256), 0, stream>>>(
      Hh2, sc2, sh2, (void*)X3F, BATCH * H2DIM, H2DIM);

  // 7) head -> out [6, 8192]
  k_head<<<dim3((BATCH + 255) / 256), dim3(256), 0, stream>>>(X3F, hw, hb, out, BATCH);
}
